// STFTR_89000312308209
// MI455X (gfx1250) — matrix operand order repaired and checked by reading, not compiled, not run on hardware
//
#include <hip/hip_runtime.h>
#include <hip/hip_bf16.h>
#include <math.h>

// ---------------------------------------------------------------------------
// Dims (from reference): S=3, B=2, N=2048, K=16, M=32, C=256
// Algebraic restructure:
//   temporal:  logits = (q0 @ (Wq Wk^T) + Wk bq) . kv   (bk/bq const terms cancel)
//              out    = sum_s (sum_k attn_k kv_k) @ (Wv Wo) + sum_s(bv Wo + bo)
//   cross:     logits = sf . ((out Wq_c + bq_c) @ Wk_c^T)
//              cs     = (sum_m a_m sf_m) @ (Wv_c Wo_c) + bv_c Wo_c + bo_c
// => heavy kernels are pure streaming (HBM-bound, ~540MB @ 23.3TB/s),
//    GEMMs are small 4096x256x256 done with v_wmma_f32_16x16x32_f16.
// ---------------------------------------------------------------------------

#define DS 3
#define DB 2
#define DN 2048
#define DK 16
#define DM 32
#define DC 256
#define BN (DB * DN)      // 4096
#define SBN (DS * BN)     // 12288
#define CC (DC * DC)      // 65536

typedef __attribute__((ext_vector_type(16))) _Float16 v16h;
typedef __attribute__((ext_vector_type(8)))  float    v8f;
typedef __attribute__((ext_vector_type(4)))  float    fvec4;

// ------------------------------ helpers -----------------------------------

__device__ __forceinline__ float wave_red_sum(float v) {
#pragma unroll
  for (int off = 16; off > 0; off >>= 1) v += __shfl_xor(v, off, 32);
  return v;
}

// ------------------------- prep: transposes & biases -----------------------

__global__ void transpose256(const float* __restrict__ in, float* __restrict__ out) {
  const float* src = in  + (long)blockIdx.z * CC;
  float*       dst = out + (long)blockIdx.z * CC;
  int idx = blockIdx.x * 256 + threadIdx.x;   // 256 blocks x 256 threads = 65536
  int i = idx >> 8, j = idx & 255;
  dst[j * DC + i] = src[idx];
}

__global__ void bias_prep(const float* __restrict__ Wk_t, const float* __restrict__ bq_t,
                          const float* __restrict__ Wk_c, const float* __restrict__ bq_c,
                          const float* __restrict__ bv_t, const float* __restrict__ Wo_t,
                          const float* __restrict__ bo_t,
                          const float* __restrict__ bv_c, const float* __restrict__ Wo_c,
                          const float* __restrict__ bo_c,
                          float* __restrict__ w1, float* __restrict__ beffc,
                          float* __restrict__ cb1, float* __restrict__ cb2) {
  int c = threadIdx.x;
  // w1[s][c] = sum_d Wk_t[s][c][d] * bq_t[s][d]
  for (int s = 0; s < DS; ++s) {
    float a = 0.f;
    for (int d = 0; d < DC; ++d) a += Wk_t[(long)s * CC + c * DC + d] * bq_t[s * DC + d];
    w1[s * DC + c] = a;
  }
  { // beffc[c] = sum_d Wk_c[c][d] * bq_c[d]
    float a = 0.f;
    for (int d = 0; d < DC; ++d) a += Wk_c[c * DC + d] * bq_c[d];
    beffc[c] = a;
  }
  { // cb1[c] = sum_s ( sum_d bv_t[s][d]*Wo_t[s][d][c] + bo_t[s][c] )
    float a = 0.f;
    for (int s = 0; s < DS; ++s) {
      for (int d = 0; d < DC; ++d) a += bv_t[s * DC + d] * Wo_t[(long)s * CC + d * DC + c];
      a += bo_t[s * DC + c];
    }
    cb1[c] = a;
  }
  { // cb2[c] = sum_d bv_c[d]*Wo_c[d][c] + bo_c[c]
    float a = 0.f;
    for (int d = 0; d < DC; ++d) a += bv_c[d] * Wo_c[d * DC + c];
    cb2[c] = a + bo_c[c];
  }
}

// ------------------------- WMMA NT-GEMM ------------------------------------
// D[i,j] = bias[j] + sum_seg sum_k A[i,k] * Bsrc[j,k]   (dot of rows, "NT")
// A: f32 rows (optionally A+A2), Bsrc: f32 or f16 rows, D: f32 or f16.
// One wave computes one 16x16 tile with v_wmma_f32_16x16x32_f16.
// VGPR layouts used here (contiguous k order, same for A and B):
//   A  (16x32 f16): lane m=l&15, hi=l>>4; elems 0..15 -> K=k0+16*hi+e
//   B  (32x16 f16): lane n=l&15, hi=l>>4; elems 0..15 -> K=k0+16*hi+e
//   C/D(16x16 f32): elem r -> row 8*hi+r, col l&15

template <typename BT, typename OT, bool DUAL_A>
__global__ __launch_bounds__(256)
void gemm_nt(const float* __restrict__ A, const float* __restrict__ A2,
             const BT* __restrict__ Bm, OT* __restrict__ D,
             const float* __restrict__ bias,
             int M, int N, int Kd, int nseg,
             long sa, long sb,                    // per-segment strides (elems)
             long ba, long bb, long bd, long bbias) { // per-batch (gridDim.z) strides
  const int lane = threadIdx.x & 31;
  const int m16 = lane & 15, hi = lane >> 4;
  const int tilesN = N >> 4;
  const int wid = blockIdx.x * blockDim.y + threadIdx.y;
  if (wid >= (M >> 4) * tilesN) return;
  const int tm = wid / tilesN, tn = wid % tilesN;
  const long z = blockIdx.z;

  const float* Az  = A + z * ba;
  const float* A2z = DUAL_A ? (A2 + z * ba) : nullptr;
  const BT*    Bz  = Bm + z * bb;
  OT*          Dz  = D + z * bd;

  v8f acc;
  {
    float b0 = bias ? bias[z * bbias + tn * 16 + m16] : 0.f;
#pragma unroll
    for (int r = 0; r < 8; ++r) acc[r] = b0;
  }

  for (int seg = 0; seg < nseg; ++seg) {
    const float* Arow = Az + seg * sa + (long)(tm * 16 + m16) * Kd;
    const float* A2row = nullptr;
    if constexpr (DUAL_A) A2row = A2z + seg * sa + (long)(tm * 16 + m16) * Kd;
    const BT* Brow = Bz + seg * sb + (long)(tn * 16 + m16) * Kd;

    for (int k0 = 0; k0 < Kd; k0 += 32) {
      v16h af;
      {
        const float* p0 = Arow + k0 + 16 * hi;
        const float* p1 = p0 + 8;
        fvec4 x0 = *(const fvec4*)(p0);
        fvec4 x1 = *(const fvec4*)(p0 + 4);
        fvec4 x2 = *(const fvec4*)(p1);
        fvec4 x3 = *(const fvec4*)(p1 + 4);
        if constexpr (DUAL_A) {
          const float* q0 = A2row + k0 + 16 * hi;
          const float* q1 = q0 + 8;
          x0 += *(const fvec4*)(q0);
          x1 += *(const fvec4*)(q0 + 4);
          x2 += *(const fvec4*)(q1);
          x3 += *(const fvec4*)(q1 + 4);
        }
#pragma unroll
        for (int j = 0; j < 4; ++j) {
          af[j]      = (_Float16)x0[j];
          af[4 + j]  = (_Float16)x1[j];
          af[8 + j]  = (_Float16)x2[j];
          af[12 + j] = (_Float16)x3[j];
        }
      }
      v16h bf;
      if constexpr (sizeof(BT) == 2) {
        bf = *(const v16h*)(Brow + k0 + 16 * hi);
      } else {
        const float* p0 = (const float*)(Brow + k0 + 16 * hi);
        fvec4 y0 = *(const fvec4*)(p0);
        fvec4 y1 = *(const fvec4*)(p0 + 4);
        fvec4 y2 = *(const fvec4*)(p0 + 8);
        fvec4 y3 = *(const fvec4*)(p0 + 12);
#pragma unroll
        for (int j = 0; j < 4; ++j) {
          bf[j]      = (_Float16)y0[j];
          bf[4 + j]  = (_Float16)y1[j];
          bf[8 + j]  = (_Float16)y2[j];
          bf[12 + j] = (_Float16)y3[j];
        }
      }
      acc = __builtin_amdgcn_wmma_f32_16x16x32_f16(false, af, false, bf,
                                                   (short)0, acc, false, false);
    }
  }

  OT* dp = Dz + (long)(tm * 16 + 8 * hi) * N + tn * 16 + m16;
#pragma unroll
  for (int r = 0; r < 8; ++r) dp[(long)r * N] = (OT)acc[r];
}

// ---------------------- temporal attention (stream kv once) ----------------
// Per query g in [0, S*B*N): kv tile (16x256 f32) -> LDS; dots vs qeff row;
// softmax over K=16; avkv = sum_k p_k * kv_k. 4 queries/block, 1 wave each.

__global__ __launch_bounds__(128)
void attn_temporal(const float* __restrict__ knn_feat,
                   const float* __restrict__ knn_pe,
                   const float* __restrict__ qeff,
                   float* __restrict__ avkv) {
  __shared__ __align__(16) float kv[4 * DK * DC];   // 64 KB
  const int tid = threadIdx.x;
  const long g0 = (long)blockIdx.x * 4;
  const fvec4* f4 = (const fvec4*)(knn_feat + g0 * DK * DC);
  const fvec4* p4 = (const fvec4*)(knn_pe   + g0 * DK * DC);
  fvec4* s4 = (fvec4*)kv;
#pragma unroll
  for (int i = 0; i < 32; ++i) {
    int idx = tid + 128 * i;
    s4[idx] = f4[idx] + p4[idx];
  }
  __syncthreads();

  const int w = tid >> 5, lane = tid & 31;
  const long g = g0 + w;
  float qr[8];
  {
    const float* q = qeff + g * DC;
#pragma unroll
    for (int j = 0; j < 8; ++j) qr[j] = q[lane + 32 * j];
  }
  const float* kvq = kv + w * (DK * DC);
  float dot[DK];
#pragma unroll
  for (int k = 0; k < DK; ++k) {
    const float* row = kvq + k * DC;
    float p = 0.f;
#pragma unroll
    for (int j = 0; j < 8; ++j) p += qr[j] * row[lane + 32 * j];
    dot[k] = wave_red_sum(p);   // full sum in every lane
  }
  const float scale = 0.0625f;  // 1/sqrt(256)
  float mx = dot[0];
#pragma unroll
  for (int k = 1; k < DK; ++k) mx = fmaxf(mx, dot[k]);
  float sum = 0.f;
#pragma unroll
  for (int k = 0; k < DK; ++k) { dot[k] = expf((dot[k] - mx) * scale); sum += dot[k]; }
  const float rinv = 1.f / sum;
  float acc[8];
#pragma unroll
  for (int j = 0; j < 8; ++j) acc[j] = 0.f;
#pragma unroll
  for (int k = 0; k < DK; ++k) {
    const float wk = dot[k] * rinv;
    const float* row = kvq + k * DC;
#pragma unroll
    for (int j = 0; j < 8; ++j) acc[j] += wk * row[lane + 32 * j];
  }
  float* o = avkv + g * DC;
#pragma unroll
  for (int j = 0; j < 8; ++j) o[lane + 32 * j] = acc[j];
}

// ---------------------- cross-sensor attention -----------------------------
// Per query bn: sampled tile (32x256 f32) -> LDS; masked softmax over M=32.

__global__ __launch_bounds__(64)
void attn_cross(const float* __restrict__ sampled,
                const float* __restrict__ qk2,
                const unsigned char* __restrict__ mask,
                float* __restrict__ av2) {
  __shared__ __align__(16) float sf[2 * DM * DC];   // 64 KB
  const int tid = threadIdx.x;
  const long g0 = (long)blockIdx.x * 2;
  const fvec4* src = (const fvec4*)(sampled + g0 * DM * DC);
  fvec4* dst = (fvec4*)sf;
#pragma unroll
  for (int i = 0; i < 64; ++i) {
    int idx = tid + 64 * i;
    dst[idx] = src[idx];
  }
  __syncthreads();

  const int w = tid >> 5, lane = tid & 31;
  const long g = g0 + w;
  float qr[8];
  {
    const float* q = qk2 + g * DC;
#pragma unroll
    for (int j = 0; j < 8; ++j) qr[j] = q[lane + 32 * j];
  }
  const float* Sq = sf + w * (DM * DC);
  float dot[DM];
#pragma unroll
  for (int m = 0; m < DM; ++m) {
    const float* row = Sq + m * DC;
    float p = 0.f;
#pragma unroll
    for (int j = 0; j < 8; ++j) p += qr[j] * row[lane + 32 * j];
    dot[m] = wave_red_sum(p);
  }
  const unsigned vmask = (unsigned)__ballot(mask[g * DM + lane] != 0);
  const float scale = 0.0625f;
  float mx = -3.4e38f;
#pragma unroll
  for (int m = 0; m < DM; ++m) if ((vmask >> m) & 1u) mx = fmaxf(mx, dot[m]);
  float sum = 0.f;
#pragma unroll
  for (int m = 0; m < DM; ++m) {
    float p = ((vmask >> m) & 1u) ? expf((dot[m] - mx) * scale) : 0.f;
    dot[m] = p; sum += p;
  }
  const float rinv = 1.f / sum;
  float acc[8];
#pragma unroll
  for (int j = 0; j < 8; ++j) acc[j] = 0.f;
#pragma unroll
  for (int m = 0; m < DM; ++m) {
    const float wm = dot[m] * rinv;
    const float* row = Sq + m * DC;
#pragma unroll
    for (int j = 0; j < 8; ++j) acc[j] += wm * row[lane + 32 * j];
  }
  float* o = av2 + g * DC;
#pragma unroll
  for (int j = 0; j < 8; ++j) o[lane + 32 * j] = acc[j];
}

// ------------------------------- launcher ----------------------------------

extern "C" void kernel_launch(void* const* d_in, const int* in_sizes, int n_in,
                              void* d_out, int out_size, void* d_ws, size_t ws_size,
                              hipStream_t stream) {
  const float* pts_feats = (const float*)d_in[0];
  const float* pts_pe    = (const float*)d_in[1];
  const float* knn_feat  = (const float*)d_in[2];
  const float* knn_pe    = (const float*)d_in[3];
  const float* sampled   = (const float*)d_in[4];
  const float* Wq_t = (const float*)d_in[5];
  const float* Wk_t = (const float*)d_in[6];
  const float* Wv_t = (const float*)d_in[7];
  const float* Wo_t = (const float*)d_in[8];
  const float* bq_t = (const float*)d_in[9];
  /* bk_t (d_in[10]) is softmax-shift-invariant -> unused */
  const float* bv_t = (const float*)d_in[11];
  const float* bo_t = (const float*)d_in[12];
  const float* Wq_c = (const float*)d_in[13];
  const float* Wk_c = (const float*)d_in[14];
  const float* Wv_c = (const float*)d_in[15];
  const float* Wo_c = (const float*)d_in[16];
  const float* bq_c = (const float*)d_in[17];
  /* bk_c (d_in[18]) unused (cancels in softmax) */
  const float* bv_c = (const float*)d_in[19];
  const float* bo_c = (const float*)d_in[20];
  const unsigned char* mask = (const unsigned char*)d_in[21];

  char* ws = (char*)d_ws;
  size_t off = 0;
  auto alloc = [&](size_t bytes) -> void* {
    void* p = ws + off;
    off += (bytes + 255) & ~(size_t)255;
    return p;
  };
  _Float16* WeffT  = (_Float16*)alloc((size_t)DS * CC * 2); // (Wq Wk^T)^T per s
  _Float16* WvoT   = (_Float16*)alloc((size_t)DS * CC * 2); // (Wv Wo)^T  per s
  _Float16* WeffTc = (_Float16*)alloc((size_t)CC * 2);
  _Float16* WvoTc  = (_Float16*)alloc((size_t)CC * 2);
  float* WoTt  = (float*)alloc((size_t)DS * CC * 4);
  float* WoTc  = (float*)alloc((size_t)CC * 4);
  float* w1    = (float*)alloc((size_t)DS * DC * 4);
  float* beffc = (float*)alloc((size_t)DC * 4);
  float* cb1   = (float*)alloc((size_t)DC * 4);
  float* cb2   = (float*)alloc((size_t)DC * 4);
  float* qeff  = (float*)alloc((size_t)DS * BN * DC * 4);
  float* avkv  = (float*)alloc((size_t)DS * BN * DC * 4);
  float* outb  = (float*)alloc((size_t)BN * DC * 4);
  float* qk2   = (float*)alloc((size_t)BN * DC * 4);
  float* av2   = (float*)alloc((size_t)BN * DC * 4);

  // 1) folded biases + Wo transposes
  bias_prep<<<1, 256, 0, stream>>>(Wk_t, bq_t, Wk_c, bq_c, bv_t, Wo_t, bo_t,
                                   bv_c, Wo_c, bo_c, w1, beffc, cb1, cb2);
  transpose256<<<dim3(256, 1, DS), 256, 0, stream>>>(Wo_t, WoTt);
  transpose256<<<dim3(256, 1, 1),  256, 0, stream>>>(Wo_c, WoTc);

  dim3 blk(32, 8);
  // 2) composed weights (f16), via WMMA
  gemm_nt<float, _Float16, false><<<dim3(32, 1, DS), blk, 0, stream>>>(
      Wk_t, nullptr, Wq_t, WeffT, nullptr, DC, DC, DC, 1, 0, 0, CC, CC, CC, 0);
  gemm_nt<float, _Float16, false><<<dim3(32, 1, DS), blk, 0, stream>>>(
      WoTt, nullptr, Wv_t, WvoT, nullptr, DC, DC, DC, 1, 0, 0, CC, CC, CC, 0);
  gemm_nt<float, _Float16, false><<<dim3(32, 1, 1), blk, 0, stream>>>(
      Wk_c, nullptr, Wq_c, WeffTc, nullptr, DC, DC, DC, 1, 0, 0, 0, 0, 0, 0);
  gemm_nt<float, _Float16, false><<<dim3(32, 1, 1), blk, 0, stream>>>(
      WoTc, nullptr, Wv_c, WvoTc, nullptr, DC, DC, DC, 1, 0, 0, 0, 0, 0, 0);

  // 3) qeff[s] = (pts_feats + pts_pe) @ Weff[s] + w1[s]
  gemm_nt<_Float16, float, true><<<dim3(512, 1, DS), blk, 0, stream>>>(
      pts_feats, pts_pe, WeffT, qeff, w1, BN, DC, DC, 1, 0, 0,
      0 /*same q0 all s*/, CC, (long)BN * DC, DC);

  // 4) temporal attention (streams 402MB of kv once; HBM-bound)
  attn_temporal<<<SBN / 4, 128, 0, stream>>>(knn_feat, knn_pe, qeff, avkv);

  // 5) out = sum_s avkv[s] @ Wvo[s] + cb1
  gemm_nt<_Float16, float, false><<<dim3(512, 1, 1), blk, 0, stream>>>(
      avkv, nullptr, WvoT, outb, cb1, BN, DC, DC, DS, (long)BN * DC, CC, 0, 0, 0, 0);

  // 6) qk2 = out @ Weff_c + beff_c
  gemm_nt<_Float16, float, false><<<dim3(512, 1, 1), blk, 0, stream>>>(
      outb, nullptr, WeffTc, qk2, beffc, BN, DC, DC, 1, 0, 0, 0, 0, 0, 0);

  // 7) cross attention (streams 134MB of sampled_feat once)
  attn_cross<<<BN / 2, 64, 0, stream>>>(sampled, qk2, mask, av2);

  // 8) cs = av2 @ Wvo_c + cb2  -> d_out
  gemm_nt<_Float16, float, false><<<dim3(512, 1, 1), blk, 0, stream>>>(
      av2, nullptr, WvoTc, (float*)d_out, cb2, BN, DC, DC, 1, 0, 0, 0, 0, 0, 0);
}
